// EvaSparseLinearAttention_20890720928475
// MI455X (gfx1250) — matrix operand order repaired and checked by reading, not compiled, not run on hardware
//
#include <hip/hip_runtime.h>
#include <hip/hip_bf16.h>

// MI455X / gfx1250, wave32. All matrix math via V_WMMA_F32_16X16X32_BF16.
// All WMMA operands pre-staged in bf16 in fragment-friendly layouts so that
// A-frags = 2x b128 loads/lane, B-frags = 2x b128 loads/lane, no cvt in loops.
//
// Shapes: B=2, N=2048, C=1024, H=16, D=64, 64-blocks, nqb=nkb=32, top-8.

#define Bq 2
#define Nn 2048
#define Cc 1024
#define Hh 16
#define Dd 64
#define NB 32
#define KC 8
#define BHND ((size_t)Bq * Hh * Nn * Dd)   // 4194304

typedef __attribute__((ext_vector_type(16))) __bf16 bf16x16;
typedef __attribute__((ext_vector_type(8)))  __bf16 bf16x8;
typedef __attribute__((ext_vector_type(8)))  float  floatx8;

__device__ __forceinline__ __bf16 f2bf(float f) {
  unsigned u = __builtin_bit_cast(unsigned, f);
  unsigned r = (u + 0x7FFFu + ((u >> 16) & 1u)) >> 16;   // RNE
  unsigned short s = (unsigned short)r;
  return __builtin_bit_cast(__bf16, s);
}
__device__ __forceinline__ float bf2f(__bf16 b) {
  unsigned u = (unsigned)__builtin_bit_cast(unsigned short, b) << 16;
  return __builtin_bit_cast(float, u);
}

__device__ __forceinline__ floatx8 wmma_bf16(bf16x16 a, bf16x16 b, floatx8 c) {
  return __builtin_amdgcn_wmma_f32_16x16x32_bf16(false, a, false, b, (short)0, c,
                                                 false, false);
}

// ---- packed fragment loaders (ISA 7.12.2, wave32) -------------------------
// A 16x32 (MxK) from bf16 row-major [M x K]; lane holds row lane&15,
// K = half*8 + 0..7 and 16 + half*8 + 0..7  ->  two contiguous 16B loads.
__device__ __forceinline__ bf16x16 load_a_bf(const __bf16* p, int ld) {
  int lane = threadIdx.x & 31;
  const __bf16* rp = p + (size_t)(lane & 15) * ld + ((lane >> 4) * 8);
  bf16x8 lo = *(const bf16x8*)rp;
  bf16x8 hi = *(const bf16x8*)(rp + 16);
  return __builtin_shufflevector(lo, hi, 0, 1, 2, 3, 4, 5, 6, 7,
                                 8, 9, 10, 11, 12, 13, 14, 15);
}
// B 32x16 (KxN) from bf16 [N x K] row-major (B^T storage); lane holds col
// n=lane&15, K = half*8 + 0..7 and 16 + half*8 + 0..7  ->  two 16B loads.
__device__ __forceinline__ bf16x16 load_b_bf(const __bf16* p, int ld) {
  int lane = threadIdx.x & 31;
  const __bf16* rp = p + (size_t)(lane & 15) * ld + ((lane >> 4) * 8);
  bf16x8 lo = *(const bf16x8*)rp;
  bf16x8 hi = *(const bf16x8*)(rp + 16);
  return __builtin_shufflevector(lo, hi, 0, 1, 2, 3, 4, 5, 6, 7,
                                 8, 9, 10, 11, 12, 13, 14, 15);
}
// C/D 16x16 f32: vgpr r -> row r + 8*(lane>=16), col = lane&15
__device__ __forceinline__ void store_c(float* p, int ld, floatx8 c) {
  int lane = threadIdx.x & 31;
  int col = lane & 15, rb = (lane >> 4) * 8;
#pragma unroll
  for (int r = 0; r < 8; ++r) p[(size_t)(rb + r) * ld + col] = c[r];
}

// ===========================================================================
// 0) f32 -> bf16 bulk conversion (4 contiguous elems / thread)
__global__ void k_cvt(const float* __restrict__ src, __bf16* __restrict__ dst) {
  size_t i = ((size_t)blockIdx.x * 256 + threadIdx.x) * 4;
#pragma unroll
  for (int e = 0; e < 4; ++e) dst[i + e] = f2bf(src[i + e]);
}

// 1) QKV GEMM, 2x2 register blocking (32x32 out / wave).
//    q,k -> bf16 [B,H,N,D]; v -> bf16 block-transposed [B,H,NB,D,64].
__device__ __forceinline__ void qkv_store_tile(floatx8 acc, int row0, int oc0,
                                               const float* __restrict__ qbias,
                                               const float* __restrict__ vbias,
                                               __bf16* __restrict__ qb,
                                               __bf16* __restrict__ kb,
                                               __bf16* __restrict__ vt) {
  int lane = threadIdx.x & 31;
  int col = lane & 15, rb = (lane >> 4) * 8;
  int oc = oc0 + col;
  int which = oc >> 10;                 // 0=q 1=k 2=v
  int cc = oc & (Cc - 1);
  int h = cc >> 6, d = cc & 63;
  float bias = (which == 0) ? qbias[cc] : ((which == 2) ? vbias[cc] : 0.f);
#pragma unroll
  for (int r = 0; r < 8; ++r) {
    int row = row0 + rb + r;
    int b = row >> 11, n = row & (Nn - 1);
    size_t bh = (size_t)b * Hh + h;
    float val = acc[r] + bias;
    if (which == 0)      qb[(bh * Nn + n) * Dd + d] = f2bf(val);
    else if (which == 1) kb[(bh * Nn + n) * Dd + d] = f2bf(val);
    else                 vt[((bh * NB + (n >> 6)) * Dd + d) * 64 + (n & 63)] = f2bf(val);
  }
}
__global__ void k_qkv(const __bf16* __restrict__ xb, const __bf16* __restrict__ wb,
                      const float* __restrict__ qbias, const float* __restrict__ vbias,
                      __bf16* __restrict__ qb, __bf16* __restrict__ kb,
                      __bf16* __restrict__ vt) {
  int row0 = blockIdx.x * 32, oc0 = blockIdx.y * 32;
  const __bf16* a0p = xb + (size_t)row0 * Cc;
  const __bf16* a1p = xb + (size_t)(row0 + 16) * Cc;
  const __bf16* b0p = wb + (size_t)oc0 * Cc;
  const __bf16* b1p = wb + (size_t)(oc0 + 16) * Cc;
  floatx8 acc00 = {}, acc01 = {}, acc10 = {}, acc11 = {};
  for (int kk = 0; kk < Cc; kk += 32) {
    __builtin_prefetch(a0p + kk + 256, 0, 0);
    __builtin_prefetch(b0p + kk + 256, 0, 0);
    bf16x16 a0 = load_a_bf(a0p + kk, Cc);
    bf16x16 a1 = load_a_bf(a1p + kk, Cc);
    bf16x16 b0 = load_b_bf(b0p + kk, Cc);
    bf16x16 b1 = load_b_bf(b1p + kk, Cc);
    acc00 = wmma_bf16(a0, b0, acc00);
    acc01 = wmma_bf16(a0, b1, acc01);
    acc10 = wmma_bf16(a1, b0, acc10);
    acc11 = wmma_bf16(a1, b1, acc11);
  }
  qkv_store_tile(acc00, row0, oc0, qbias, vbias, qb, kb, vt);
  qkv_store_tile(acc01, row0, oc0 + 16, qbias, vbias, qb, kb, vt);
  qkv_store_tile(acc10, row0 + 16, oc0, qbias, vbias, qb, kb, vt);
  qkv_store_tile(acc11, row0 + 16, oc0 + 16, qbias, vbias, qb, kb, vt);
}

// 2) block pooling from bf16 row-major [rows x D]
__global__ void k_pool_bf(const __bf16* __restrict__ src, float* __restrict__ dst,
                          float scl) {
  int bhi = blockIdx.x, d = threadIdx.x;
  const __bf16* p = src + (size_t)bhi * 64 * Dd + d;
  float s = 0.f;
  for (int j = 0; j < 64; ++j) s += bf2f(p[(size_t)j * Dd]);
  dst[(size_t)bhi * Dd + d] = s * scl;
}
// pooling over trailing dim of block-transposed bf16 [blk, D, 64] (Zb)
__global__ void k_poolT_bf(const __bf16* __restrict__ src, float* __restrict__ dst) {
  int bhi = blockIdx.x, d = threadIdx.x;
  const __bf16* p = src + (size_t)bhi * 4096 + (size_t)d * 64;
  float s = 0.f;
  for (int t = 0; t < 64; ++t) s += bf2f(p[t]);
  dst[(size_t)bhi * Dd + d] = s;
}

// 3) top-8 block selection + complement mask (1 thread per (b,h,i))
__global__ void k_topk(const float* __restrict__ qbm, const float* __restrict__ kbm,
                       int* __restrict__ idx, float* __restrict__ comp) {
  int t = blockIdx.x * blockDim.x + threadIdx.x;
  if (t >= Bq * Hh * NB) return;
  int bh = t >> 5;
  const float* qv = qbm + (size_t)t * Dd;
  float sc[NB];
  for (int j = 0; j < NB; ++j) {
    const float* kv = kbm + ((size_t)bh * NB + j) * Dd;
    float s = 0.f;
    for (int d = 0; d < Dd; ++d) s += qv[d] * kv[d];
    sc[j] = s * 0.125f;
  }
  unsigned sel = 0;
  for (int r = 0; r < KC; ++r) {
    int best = 0; float bv = -3.4e38f;
    for (int j = 0; j < NB; ++j) {
      if ((sel >> j) & 1u) continue;
      if (sc[j] > bv) { bv = sc[j]; best = j; }
    }
    sel |= 1u << best;
    idx[(size_t)t * KC + r] = best;
  }
  for (int j = 0; j < NB; ++j) comp[(size_t)t * NB + j] = ((sel >> j) & 1u) ? 0.f : 1.f;
}

// 4) row softmax over D from bf16; mode 0: row-major out, mode 1: out
//    block-transposed [blk, D, 64] (for phi_k as WMMA-A = phi_k^T).
__global__ void k_phi(const __bf16* __restrict__ src, __bf16* __restrict__ dst,
                      int transposed) {
  __shared__ float red[64];
  size_t row = blockIdx.x;
  int d = threadIdx.x;
  float x = bf2f(src[row * Dd + d]);
  red[d] = x; __syncthreads();
  for (int off = 32; off > 0; off >>= 1) {
    if (d < off) red[d] = fmaxf(red[d], red[d + off]);
    __syncthreads();
  }
  float m = red[0]; __syncthreads();
  float e = __expf(x - m);
  red[d] = e; __syncthreads();
  for (int off = 32; off > 0; off >>= 1) {
    if (d < off) red[d] += red[d + off];
    __syncthreads();
  }
  float r = e / red[0];
  if (transposed) {
    size_t blk = row >> 6;
    dst[blk * 4096 + (size_t)d * 64 + (row & 63)] = f2bf(r);
  } else {
    dst[row * Dd + d] = f2bf(r);
  }
}

// 5) KVb[bhj] = phik_blk^T @ v_blk (both pre-transposed bf16). f32 out.
__global__ void k_kvb(const __bf16* __restrict__ phikt, const __bf16* __restrict__ vt,
                      float* __restrict__ KVb) {
  int bhj = blockIdx.x, t = blockIdx.y;
  int d0 = (t >> 2) * 16, e0 = (t & 3) * 16;
  const __bf16* pk = phikt + (size_t)bhj * 4096;   // [d][t]
  const __bf16* pv = vt + (size_t)bhj * 4096;      // [e][t]
  floatx8 acc = {};
#pragma unroll
  for (int kk = 0; kk < 64; kk += 32) {
    bf16x16 a = load_a_bf(pk + (size_t)d0 * 64 + kk, 64);
    bf16x16 b = load_b_bf(pv + (size_t)e0 * 64 + kk, 64);
    acc = wmma_bf16(a, b, acc);
  }
  store_c(KVb + (size_t)bhj * 4096 + (size_t)d0 * Dd + e0, Dd, acc);
}

// 6) KVi = sum_j comp[i,j]*KVb[j]  -> bf16 transposed [e][d]; Zi f32.
__global__ void k_kvi(const float* __restrict__ KVb, const float* __restrict__ Zb,
                      const float* __restrict__ comp, __bf16* __restrict__ KVit,
                      float* __restrict__ Zi) {
  __shared__ float cm[NB];
  int bhi = blockIdx.x, tid = threadIdx.x;
  int bh = bhi >> 5;
  if (tid < NB) cm[tid] = comp[(size_t)bhi * NB + tid];
  __syncthreads();
  const float* kb = KVb + (size_t)bh * NB * 4096;
#pragma unroll
  for (int e = 0; e < 16; ++e) {
    int elem = tid + e * 256;                 // = d*64 + ec
    float acc = 0.f;
    for (int j = 0; j < NB; ++j) acc += cm[j] * kb[(size_t)j * 4096 + elem];
    int d = elem >> 6, ec = elem & 63;
    KVit[(size_t)bhi * 4096 + (size_t)ec * 64 + d] = f2bf(acc);
  }
  if (tid < Dd) {
    const float* zb = Zb + (size_t)bh * NB * Dd;
    float acc = 0.f;
    for (int j = 0; j < NB; ++j) acc += cm[j] * zb[(size_t)j * Dd + tid];
    Zi[(size_t)bhi * Dd + tid] = acc;
  }
}

// 7) lin_num[bhi] = phiq_blk @ KVi[bhi]   (B operand from KVit [e][d])
__global__ void k_linnum(const __bf16* __restrict__ phiq, const __bf16* __restrict__ KVit,
                         float* __restrict__ lin_num) {
  int bhi = blockIdx.x, t = blockIdx.y;
  int q0 = (t >> 2) * 16, e0 = (t & 3) * 16;
  const __bf16* pq = phiq + ((size_t)bhi * 64 + q0) * Dd;
  const __bf16* kv = KVit + (size_t)bhi * 4096 + (size_t)e0 * 64;
  floatx8 acc = {};
#pragma unroll
  for (int kk = 0; kk < 64; kk += 32) {
    bf16x16 a = load_a_bf(pq + kk, Dd);
    bf16x16 b = load_b_bf(kv + kk, 64);
    acc = wmma_bf16(a, b, acc);
  }
  store_c(lin_num + (size_t)bhi * 64 * Dd + (size_t)q0 * Dd + e0, Dd, acc);
}

// 8) lin_den[bhi*64+q] = dot(phiq[row], Zi[bhi])
__global__ void k_linden(const __bf16* __restrict__ phiq, const float* __restrict__ Zi,
                         float* __restrict__ lin_den) {
  int bhi = blockIdx.x, qi = threadIdx.x;
  const __bf16* pq = phiq + ((size_t)bhi * 64 + qi) * Dd;
  const float* z = Zi + (size_t)bhi * Dd;
  float s = 0.f;
  for (int d = 0; d < Dd; ++d) s += bf2f(pq[d]) * z[d];
  lin_den[(size_t)bhi * 64 + qi] = s;
}

// 9) sparse exact-attention + combine. grid B*H*NB*4 (16-row strips),
//    block 128 (4 waves, one 16-wide e-tile each). E staged in LDS as bf16.
__global__ void k_sparse(const __bf16* __restrict__ qb, const __bf16* __restrict__ kb,
                         const __bf16* __restrict__ vt, const int* __restrict__ idx,
                         const float* __restrict__ lin_num,
                         const float* __restrict__ lin_den, float* __restrict__ o) {
  __shared__ __bf16 Ebuf[16 * 64];
  __shared__ float den[16];
  int blk = blockIdx.x;
  int qt = blk & 3, i = (blk >> 2) & 31, bh = blk >> 7;
  int tid = threadIdx.x, wave = tid >> 5, lane = tid & 31;
  int col = lane & 15, rb = (lane >> 4) * 8;
  int e0 = wave * 16;
  int qrow0 = (bh * NB + i) * 64 + qt * 16;
  const __bf16* qbase = qb + (size_t)qrow0 * Dd;
  if (tid < 16) den[tid] = 0.f;
  __syncthreads();
  floatx8 accn = {};
  for (int s = 0; s < KC; ++s) {
    int j = idx[((size_t)bh * NB + i) * KC + s];
    const __bf16* kblk = kb + ((size_t)bh * NB + j) * 64 * Dd;   // [key][d]
    const __bf16* vblk = vt + ((size_t)bh * NB + j) * 4096;      // [e][key]
    floatx8 sacc = {};
#pragma unroll
    for (int kk = 0; kk < Dd; kk += 32) {
      bf16x16 a = load_a_bf(qbase + kk, Dd);
      bf16x16 b = load_b_bf(kblk + (size_t)e0 * Dd + kk, Dd);
      sacc = wmma_bf16(a, b, sacc);
    }
#pragma unroll
    for (int r = 0; r < 8; ++r)
      Ebuf[(rb + r) * 64 + e0 + col] = f2bf(__expf(sacc[r] * 0.125f));
    __syncthreads();
    if (tid < 16) {
      float srow = 0.f;
      for (int c = 0; c < 64; ++c) srow += bf2f(Ebuf[tid * 64 + c]);
      den[tid] += srow;
    }
#pragma unroll
    for (int kk = 0; kk < 64; kk += 32) {
      bf16x16 a = load_a_bf(&Ebuf[kk], 64);                      // ds_load_b128
      bf16x16 b = load_b_bf(vblk + (size_t)e0 * 64 + kk, 64);
      accn = wmma_bf16(a, b, accn);
    }
    __syncthreads();
  }
  const float* lnum = lin_num + (size_t)qrow0 * Dd;
  const float* lden = lin_den + qrow0;
  int b = bh >> 4, h = bh & 15;
  int nbase = i * 64 + qt * 16;
#pragma unroll
  for (int r = 0; r < 8; ++r) {
    int lr = rb + r;
    float num = accn[r] + lnum[(size_t)lr * Dd + e0 + col];
    float dd = den[lr] + lden[lr];
    o[((size_t)b * Nn + nbase + lr) * Cc + h * Dd + e0 + col] = num / dd;
  }
}

// 10) LayerNorm over C: f32 in -> bf16 out (A operand of proj)
__global__ void k_ln(const float* __restrict__ o, const float* __restrict__ g,
                     const float* __restrict__ bb, __bf16* __restrict__ ob) {
  __shared__ float red[256];
  size_t row = blockIdx.x;
  int tid = threadIdx.x;
  const float* p = o + row * Cc;
  float loc[4];
  float s = 0.f;
#pragma unroll
  for (int e = 0; e < 4; ++e) { loc[e] = p[tid + e * 256]; s += loc[e]; }
  red[tid] = s; __syncthreads();
  for (int off = 128; off > 0; off >>= 1) {
    if (tid < off) red[tid] += red[tid + off];
    __syncthreads();
  }
  float mu = red[0] * (1.f / Cc); __syncthreads();
  float s2 = 0.f;
#pragma unroll
  for (int e = 0; e < 4; ++e) { float d = loc[e] - mu; s2 += d * d; }
  red[tid] = s2; __syncthreads();
  for (int off = 128; off > 0; off >>= 1) {
    if (tid < off) red[tid] += red[tid + off];
    __syncthreads();
  }
  float rinv = rsqrtf(red[0] * (1.f / Cc) + 1e-5f);
#pragma unroll
  for (int e = 0; e < 4; ++e) {
    int c = tid + e * 256;
    ob[row * Cc + c] = f2bf((loc[e] - mu) * rinv * g[c] + bb[c]);
  }
}

// 11) output projection, 2x2 register blocking -> f32 d_out
__global__ void k_proj(const __bf16* __restrict__ ob, const __bf16* __restrict__ wb,
                       const float* __restrict__ pb, float* __restrict__ out) {
  int row0 = blockIdx.x * 32, c0 = blockIdx.y * 32;
  const __bf16* a0p = ob + (size_t)row0 * Cc;
  const __bf16* a1p = ob + (size_t)(row0 + 16) * Cc;
  const __bf16* b0p = wb + (size_t)c0 * Cc;
  const __bf16* b1p = wb + (size_t)(c0 + 16) * Cc;
  floatx8 acc00 = {}, acc01 = {}, acc10 = {}, acc11 = {};
  for (int kk = 0; kk < Cc; kk += 32) {
    __builtin_prefetch(a0p + kk + 256, 0, 0);
    __builtin_prefetch(b0p + kk + 256, 0, 0);
    bf16x16 a0 = load_a_bf(a0p + kk, Cc);
    bf16x16 a1 = load_a_bf(a1p + kk, Cc);
    bf16x16 b0 = load_b_bf(b0p + kk, Cc);
    bf16x16 b1 = load_b_bf(b1p + kk, Cc);
    acc00 = wmma_bf16(a0, b0, acc00);
    acc01 = wmma_bf16(a0, b1, acc01);
    acc10 = wmma_bf16(a1, b0, acc10);
    acc11 = wmma_bf16(a1, b1, acc11);
  }
  int lane = threadIdx.x & 31;
  int col = lane & 15, rb = (lane >> 4) * 8;
#pragma unroll
  for (int r = 0; r < 8; ++r) {
    out[(size_t)(row0 + rb + r) * Cc + c0 + col] = acc00[r] + pb[c0 + col];
    out[(size_t)(row0 + rb + r) * Cc + c0 + 16 + col] = acc01[r] + pb[c0 + 16 + col];
    out[(size_t)(row0 + 16 + rb + r) * Cc + c0 + col] = acc10[r] + pb[c0 + col];
    out[(size_t)(row0 + 16 + rb + r) * Cc + c0 + 16 + col] = acc11[r] + pb[c0 + 16 + col];
  }
}

// ===========================================================================
extern "C" void kernel_launch(void* const* d_in, const int* in_sizes, int n_in,
                              void* d_out, int out_size, void* d_ws, size_t ws_size,
                              hipStream_t stream) {
  (void)in_sizes; (void)n_in; (void)out_size; (void)ws_size;
  const float* x      = (const float*)d_in[0];
  const float* qkv_w  = (const float*)d_in[1];
  const float* q_bias = (const float*)d_in[2];
  const float* v_bias = (const float*)d_in[3];
  const float* norm_g = (const float*)d_in[4];
  const float* norm_b = (const float*)d_in[5];
  const float* proj_w = (const float*)d_in[6];
  const float* proj_b = (const float*)d_in[7];
  float* out = (float*)d_out;

  char* ws = (char*)d_ws;
  auto alloc = [&](size_t bytes) {
    void* p = (void*)ws;
    ws += (bytes + 255) & ~(size_t)255;
    return p;
  };
  __bf16* xb    = (__bf16*)alloc((size_t)Bq * Nn * Cc * 2);
  __bf16* wqb   = (__bf16*)alloc((size_t)3 * Cc * Cc * 2);
  __bf16* wpb   = (__bf16*)alloc((size_t)Cc * Cc * 2);
  __bf16* qb    = (__bf16*)alloc(BHND * 2);
  __bf16* kb    = (__bf16*)alloc(BHND * 2);
  __bf16* vt    = (__bf16*)alloc(BHND * 2);     // [B,H,NB,D,64]
  __bf16* phiq  = (__bf16*)alloc(BHND * 2);
  __bf16* phikt = (__bf16*)alloc(BHND * 2);     // [B,H,NB,D,64]
  __bf16* KVit  = (__bf16*)alloc(BHND * 2);     // [B,H,NB,e,d]
  float*  KVb   = (float*)alloc(BHND * 4);      // [B,H,NB,d,e]
  float*  qbm   = (float*)alloc((size_t)Bq * Hh * NB * Dd * 4);
  float*  kbm   = (float*)alloc((size_t)Bq * Hh * NB * Dd * 4);
  float*  Zb    = (float*)alloc((size_t)Bq * Hh * NB * Dd * 4);
  float*  Zi    = (float*)alloc((size_t)Bq * Hh * NB * Dd * 4);
  float*  lden  = (float*)alloc((size_t)Bq * Hh * Nn * 4);
  float*  comp  = (float*)alloc((size_t)Bq * Hh * NB * NB * 4);
  int*    idx   = (int*)alloc((size_t)Bq * Hh * NB * KC * 4);
  float*  o     = (float*)alloc((size_t)Bq * Nn * Cc * 4);
  float*  lnum  = KVb;   // alias: KVb dead after k_kvi, lin_num born after

  // 0) bf16 staging of x and weights
  k_cvt<<<dim3(Bq * Nn * Cc / 1024), dim3(256), 0, stream>>>(x, xb);
  k_cvt<<<dim3(3 * Cc * Cc / 1024), dim3(256), 0, stream>>>(qkv_w, wqb);
  k_cvt<<<dim3(Cc * Cc / 1024), dim3(256), 0, stream>>>(proj_w, wpb);
  // 1) QKV projection
  k_qkv<<<dim3(Bq * Nn / 32, 3 * Cc / 32), dim3(32), 0, stream>>>(
      xb, wqb, q_bias, v_bias, qb, kb, vt);
  // 2) block pooling (means)
  k_pool_bf<<<dim3(Bq * Hh * NB), dim3(64), 0, stream>>>(qb, qbm, 1.f / 64.f);
  k_pool_bf<<<dim3(Bq * Hh * NB), dim3(64), 0, stream>>>(kb, kbm, 1.f / 64.f);
  // 3) top-k selection + complement mask
  k_topk<<<dim3(4), dim3(256), 0, stream>>>(qbm, kbm, idx, comp);
  // 4) feature maps (phi_k stored block-transposed), Zb
  k_phi<<<dim3(Bq * Hh * Nn), dim3(64), 0, stream>>>(qb, phiq, 0);
  k_phi<<<dim3(Bq * Hh * Nn), dim3(64), 0, stream>>>(kb, phikt, 1);
  k_poolT_bf<<<dim3(Bq * Hh * NB), dim3(64), 0, stream>>>(phikt, Zb);
  // 5) per-block phi(k)^T v
  k_kvb<<<dim3(Bq * Hh * NB, 16), dim3(32), 0, stream>>>(phikt, vt, KVb);
  // 6) complement-masked accumulation -> KVit (bf16, transposed), Zi
  k_kvi<<<dim3(Bq * Hh * NB), dim3(256), 0, stream>>>(KVb, Zb, comp, KVit, Zi);
  // 7) linear branch numerator / denominator
  k_linnum<<<dim3(Bq * Hh * NB, 16), dim3(32), 0, stream>>>(phiq, KVit, lnum);
  k_linden<<<dim3(Bq * Hh * NB), dim3(64), 0, stream>>>(phiq, Zi, lden);
  // 8) sparse branch + combine -> o [B,N,C] f32
  k_sparse<<<dim3(Bq * Hh * NB * 4), dim3(128), 0, stream>>>(
      qb, kb, vt, idx, lnum, lden, o);
  // 9) LayerNorm -> bf16
  k_ln<<<dim3(Bq * Nn), dim3(256), 0, stream>>>(o, norm_g, norm_b, xb /*reuse*/);
  // 10) output projection
  k_proj<<<dim3(Bq * Nn / 32, Cc / 32), dim3(32), 0, stream>>>(xb, wpb, proj_b, out);
}
